// GraphAttentionLayer_48198122996249
// MI455X (gfx1250) — matrix operand order repaired and checked by reading, not compiled, not run on hardware
//
#include <hip/hip_runtime.h>
#include <hip/hip_bf16.h>

typedef __attribute__((ext_vector_type(16))) _Float16 v16h;
typedef __attribute__((ext_vector_type(8)))  _Float16 v8h;
typedef __attribute__((ext_vector_type(8)))  float    v8f;
typedef __attribute__((ext_vector_type(4)))  float    v4f;
typedef __attribute__((ext_vector_type(2)))  float    v2f;
typedef __attribute__((ext_vector_type(4)))  int      v4i;

#define GAT_ALPHA 0.2f
#define GAT_NEG_INF -9.0e15f
#define GAT_N 8192
#define GAT_FIN 256
#define GAT_FOUT 64

// ---------------------------------------------------------------------------
// Kernel 1: Wh = h @ W   [8192,256]x[256,64] in exact f32 via WMMA 16x16x4_f32
// One wave (32 lanes) per 16-row tile; 4 N-tiles of 16 cols; K stepped by 4.
// ---------------------------------------------------------------------------
__global__ __launch_bounds__(256) void gat_wh_kernel(const float* __restrict__ h,
                                                     const float* __restrict__ W,
                                                     float* __restrict__ Wh)
{
    const int lane = threadIdx.x & 31;
    const int wv   = threadIdx.x >> 5;
    const int tile = blockIdx.x * 8 + wv;      // 0..511
    const int i0   = tile * 16;
    const int col  = lane & 15;                // M for A / N for B,C
    const int hx   = lane >> 4;                // lane half

    v8f c0 = {}, c1 = {}, c2 = {}, c3 = {};

    for (int k0 = 0; k0 < GAT_FIN; k0 += 4) {
        const int k = k0 + hx * 2;
        // A operand: 16x4 f32; lane holds h[i0+col][k], h[i0+col][k+1]
        v2f av = *(const v2f*)(h + (size_t)(i0 + col) * GAT_FIN + k);
        // B operand: 4x16 f32; lane holds W[k][n], W[k+1][n] with n = t*16+col
        const float* wp = W + (size_t)k * GAT_FOUT + col;
        v2f b0, b1, b2, b3;
        b0.x = wp[0];  b0.y = wp[GAT_FOUT + 0];
        b1.x = wp[16]; b1.y = wp[GAT_FOUT + 16];
        b2.x = wp[32]; b2.y = wp[GAT_FOUT + 32];
        b3.x = wp[48]; b3.y = wp[GAT_FOUT + 48];
        c0 = __builtin_amdgcn_wmma_f32_16x16x4_f32(false, av, false, b0, (short)0, c0, false, false);
        c1 = __builtin_amdgcn_wmma_f32_16x16x4_f32(false, av, false, b1, (short)0, c1, false, false);
        c2 = __builtin_amdgcn_wmma_f32_16x16x4_f32(false, av, false, b2, (short)0, c2, false, false);
        c3 = __builtin_amdgcn_wmma_f32_16x16x4_f32(false, av, false, b3, (short)0, c3, false, false);
    }

    // C/D layout: reg r -> row r (lanes 0-15) / row r+8 (lanes 16-31), N = col
    #pragma unroll
    for (int r = 0; r < 8; ++r) {
        size_t o = (size_t)(i0 + r + hx * 8) * GAT_FOUT + col;
        Wh[o]      = c0[r];
        Wh[o + 16] = c1[r];
        Wh[o + 32] = c2[r];
        Wh[o + 48] = c3[r];
    }
}

// ---------------------------------------------------------------------------
// Kernel 2: f1 = Wh@a1, f2 = Wh@a2, and WhT[n][i] = (f16)Wh[i][n]
// ---------------------------------------------------------------------------
__global__ __launch_bounds__(256) void gat_prep_kernel(const float* __restrict__ Wh,
                                                       const float* __restrict__ a,
                                                       _Float16* __restrict__ WhT,
                                                       float* __restrict__ f1,
                                                       float* __restrict__ f2)
{
    const int i = blockIdx.x * 256 + threadIdx.x;   // 8192 threads
    const float* wr = Wh + (size_t)i * GAT_FOUT;
    float s1 = 0.f, s2 = 0.f;
    #pragma unroll
    for (int n = 0; n < GAT_FOUT; ++n) {
        float w = wr[n];
        s1 += w * a[n];
        s2 += w * a[n + GAT_FOUT];
        WhT[(size_t)n * GAT_N + i] = (_Float16)w;   // coalesced across i
    }
    f1[i] = s1;
    f2[i] = s2;
}

// ---------------------------------------------------------------------------
// Kernel 3: fused masked softmax + attention@Wh (flash-style, one adj pass)
// One wave per 16 output rows. Per 32-col chunk: masked leaky-relu scores,
// online softmax, P (f16) @ WhT tile via V_WMMA_F32_16X16X32_F16.
// A-operand f16 16x32 layout: lanes 0-15 row M, K = {0..7,16..23}+0;
// lanes 16-31 same rows, K = {8..15,24..31}. Element e -> K = hx*8+(e&7)+(e>>3)*16.
// ---------------------------------------------------------------------------
__global__ __launch_bounds__(256) void gat_attn_kernel(const int* __restrict__ adj,
                                                       const float* __restrict__ f1,
                                                       const float* __restrict__ f2,
                                                       const _Float16* __restrict__ WhT,
                                                       float* __restrict__ out)
{
    const int lane = threadIdx.x & 31;
    const int wv   = threadIdx.x >> 5;
    const int tile = blockIdx.x * 8 + wv;   // 0..511
    const int i0   = tile * 16;
    const int row  = lane & 15;
    const int hx   = lane >> 4;

    const float f1r = f1[i0 + row];
    float m = GAT_NEG_INF;
    float l = 0.f;
    v8f acc0 = {}, acc1 = {}, acc2 = {}, acc3 = {};

    const int*   arow = adj + (size_t)(i0 + row) * GAT_N + hx * 8;
    const float* f2b  = f2 + hx * 8;

    for (int j0 = 0; j0 < GAT_N; j0 += 32) {
        // adj tile: two contiguous 8-int runs per lane (matches A lane layout)
        v4i m0 = *(const v4i*)(arow + j0);
        v4i m1 = *(const v4i*)(arow + j0 + 4);
        v4i m2 = *(const v4i*)(arow + j0 + 16);
        v4i m3 = *(const v4i*)(arow + j0 + 20);
        v4f g0 = *(const v4f*)(f2b + j0);
        v4f g1 = *(const v4f*)(f2b + j0 + 4);
        v4f g2 = *(const v4f*)(f2b + j0 + 16);
        v4f g3 = *(const v4f*)(f2b + j0 + 20);

        float s[16];
        #pragma unroll
        for (int e = 0; e < 4; ++e) {
            float t0 = f1r + g0[e];
            float t1 = f1r + g1[e];
            float t2 = f1r + g2[e];
            float t3 = f1r + g3[e];
            t0 = (t0 > 0.f) ? t0 : GAT_ALPHA * t0;
            t1 = (t1 > 0.f) ? t1 : GAT_ALPHA * t1;
            t2 = (t2 > 0.f) ? t2 : GAT_ALPHA * t2;
            t3 = (t3 > 0.f) ? t3 : GAT_ALPHA * t3;
            s[e]      = (m0[e] > 0) ? t0 : GAT_NEG_INF;
            s[e + 4]  = (m1[e] > 0) ? t1 : GAT_NEG_INF;
            s[e + 8]  = (m2[e] > 0) ? t2 : GAT_NEG_INF;
            s[e + 12] = (m3[e] > 0) ? t3 : GAT_NEG_INF;
        }

        // online softmax: row max across both lane halves
        float cmax = s[0];
        #pragma unroll
        for (int e = 1; e < 16; ++e) cmax = fmaxf(cmax, s[e]);
        cmax = fmaxf(cmax, __shfl_xor(cmax, 16, 32));
        const float mnew  = fmaxf(m, cmax);
        const float scale = __expf(m - mnew);

        v16h pa;
        float psum = 0.f;
        #pragma unroll
        for (int e = 0; e < 16; ++e) {
            float p = __expf(s[e] - mnew);
            psum += p;
            pa[e] = (_Float16)p;
        }
        l = l * scale + psum;
        m = mnew;

        // rescale accumulators: reg r holds rows r (lanes 0-15) / r+8 (16-31)
        #pragma unroll
        for (int r = 0; r < 8; ++r) {
            float sr = __shfl(scale, r + hx * 8, 32);
            acc0[r] *= sr;
            acc1[r] *= sr;
            acc2[r] *= sr;
            acc3[r] *= sr;
        }

        // B operands from WhT: per lane two 8-half runs at K = hx*8 and 16+hx*8
        const size_t bo = (size_t)j0 + hx * 8;
        {
            const _Float16* bp = WhT + (size_t)(0 * 16 + row) * GAT_N + bo;
            v8h blo = *(const v8h*)bp;
            v8h bhi = *(const v8h*)(bp + 16);
            v16h bb;
            #pragma unroll
            for (int e = 0; e < 8; ++e) { bb[e] = blo[e]; bb[e + 8] = bhi[e]; }
            acc0 = __builtin_amdgcn_wmma_f32_16x16x32_f16(false, pa, false, bb, (short)0, acc0, false, false);
        }
        {
            const _Float16* bp = WhT + (size_t)(1 * 16 + row) * GAT_N + bo;
            v8h blo = *(const v8h*)bp;
            v8h bhi = *(const v8h*)(bp + 16);
            v16h bb;
            #pragma unroll
            for (int e = 0; e < 8; ++e) { bb[e] = blo[e]; bb[e + 8] = bhi[e]; }
            acc1 = __builtin_amdgcn_wmma_f32_16x16x32_f16(false, pa, false, bb, (short)0, acc1, false, false);
        }
        {
            const _Float16* bp = WhT + (size_t)(2 * 16 + row) * GAT_N + bo;
            v8h blo = *(const v8h*)bp;
            v8h bhi = *(const v8h*)(bp + 16);
            v16h bb;
            #pragma unroll
            for (int e = 0; e < 8; ++e) { bb[e] = blo[e]; bb[e + 8] = bhi[e]; }
            acc2 = __builtin_amdgcn_wmma_f32_16x16x32_f16(false, pa, false, bb, (short)0, acc2, false, false);
        }
        {
            const _Float16* bp = WhT + (size_t)(3 * 16 + row) * GAT_N + bo;
            v8h blo = *(const v8h*)bp;
            v8h bhi = *(const v8h*)(bp + 16);
            v16h bb;
            #pragma unroll
            for (int e = 0; e < 8; ++e) { bb[e] = blo[e]; bb[e + 8] = bhi[e]; }
            acc3 = __builtin_amdgcn_wmma_f32_16x16x32_f16(false, pa, false, bb, (short)0, acc3, false, false);
        }
    }

    // finalize: divide by row sum, ELU, store
    float lf = l + __shfl_xor(l, 16, 32);
    #pragma unroll
    for (int r = 0; r < 8; ++r) {
        float lr  = __shfl(lf, r + hx * 8, 32);
        float inv = 1.0f / lr;
        size_t o = (size_t)(i0 + r + hx * 8) * GAT_FOUT + row;
        float v0 = acc0[r] * inv;
        float v1 = acc1[r] * inv;
        float v2 = acc2[r] * inv;
        float v3 = acc3[r] * inv;
        out[o]      = (v0 > 0.f) ? v0 : (__expf(v0) - 1.0f);
        out[o + 16] = (v1 > 0.f) ? v1 : (__expf(v1) - 1.0f);
        out[o + 32] = (v2 > 0.f) ? v2 : (__expf(v2) - 1.0f);
        out[o + 48] = (v3 > 0.f) ? v3 : (__expf(v3) - 1.0f);
    }
}

// ---------------------------------------------------------------------------
extern "C" void kernel_launch(void* const* d_in, const int* in_sizes, int n_in,
                              void* d_out, int out_size, void* d_ws, size_t ws_size,
                              hipStream_t stream) {
    const float* h   = (const float*)d_in[0];   // [8192,256]
    const int*   adj = (const int*)d_in[1];     // [8192,8192]
    const float* W   = (const float*)d_in[2];   // [256,64]
    const float* a   = (const float*)d_in[3];   // [128,1]
    float* out = (float*)d_out;                 // [8192,64]

    char* ws = (char*)d_ws;
    float*    Wh  = (float*)ws;                              // 2 MB
    _Float16* WhT = (_Float16*)(ws + (size_t)2 * 1024 * 1024); // 1 MB
    float*    f1  = (float*)(ws + (size_t)3 * 1024 * 1024);  // 32 KB
    float*    f2  = f1 + GAT_N;                              // 32 KB

    gat_wh_kernel<<<64, 256, 0, stream>>>(h, W, Wh);
    gat_prep_kernel<<<GAT_N / 256, 256, 0, stream>>>(Wh, a, WhT, f1, f2);
    gat_attn_kernel<<<64, 256, 0, stream>>>(adj, f1, f2, WhT, out);
}
